// Model_33105607917649
// MI455X (gfx1250) — matrix operand order repaired and checked by reading, not compiled, not run on hardware
//
#include <hip/hip_runtime.h>

// ---------------------------------------------------------------------------
// MI455X (gfx1250) window-CNN, round 2.
//  - one workgroup per (b,t) window; all activations in LDS.
//  - conv1 layers use a zero halo (32x106x17 f16) so pad-1 needs no predication
//  - B-im2col gathered ONCE per (spatial tile, k-chunk), reused by MT
//    accumulator tiles -> 16/MT ds_load_u16 per v_wmma
//  - conv3 (12x9, K=6912) is K-split across the 8 waves with ds_add_f32
//    LDS reduction so all waves stay busy on the heaviest layer
//  - weights pre-swizzled to the CDNA5 A-fragment lane layout (zero padded,
//    which is what makes predication-free gathers legal)
// ---------------------------------------------------------------------------

typedef __attribute__((ext_vector_type(16))) _Float16 v16h;
typedef __attribute__((ext_vector_type(8)))  float    v8f;

#define CTX   7
#define NB    4
#define NT    512
#define NF    104
#define NWIN  15
#define NOUT  25

#define HBUF  57664        // halves per LDS buffer: 32*106*17
#define C3ACC_F32  4992    // 128*39 f32 accumulator for conv3
#define C3OUT_OFF  24576   // f16 conv3 output offset (halves) inside regA

// pre-swizzled weight offsets in d_ws (f16 elements), layout
// [mt*Kchunks + kc][lane(32)][e(16)]
#define OFF1A 0u
#define OFF1B 1024u
#define OFF1C 10240u
#define OFF1D 19456u
#define OFF2A 28672u
#define OFF2B 47104u
#define OFF3  83968u
#define OFFL  968704u

// ---------------------------------------------------------------------------
// Weight pre-swizzle: OIHW f32 -> f16 A-fragments (16x32; k order is this code's, not the ISA table's):
//   l<16 : K = e;  l>=16: K = e+16 (contiguous, matches B gather)
// Out-of-range (co,kg) -> 0, which lets the conv kernels skip B predication.
// ---------------------------------------------------------------------------
__global__ void prep_weights(const float* __restrict__ w, _Float16* __restrict__ dst,
                             int Cout, int Ktot, int Mtiles, int Kchunks) {
  int total = Mtiles * Kchunks * 512;
  for (int i = blockIdx.x * blockDim.x + threadIdx.x; i < total;
       i += gridDim.x * blockDim.x) {
    int e    = i & 15;
    int l    = (i >> 4) & 31;
    int tile = i >> 9;
    int kc   = tile % Kchunks;
    int mt   = tile / Kchunks;
    int co   = mt * 16 + (l & 15);
    int kl   = (l < 16) ? e : e + 16;
    int kg   = kc * 32 + kl;
    float v  = (co < Cout && kg < Ktot) ? w[(size_t)co * Ktot + kg] : 0.0f;
    dst[i]   = (_Float16)v;
  }
}

// ---------------------------------------------------------------------------
// Generic conv layer: wave handles spatial tiles, MT output-channel tiles
// share one gathered B fragment. No per-element predication:
//   src index = (ci*SH + h+kh)*SW + (w+kw)   (halo or pad-0 layout)
// B layout: lane%16 = N, lanes0-15 K=e, lanes16-31 K=e+16.
// C/D: VGPR r -> M = r + 8*(lane>=16).
// ---------------------------------------------------------------------------
template<int KCH, int SH, int SW, int KH, int KW, int Cout, int Hd, int Wd,
         int DHALO, int MT, bool RELU, bool OUTF32>
__device__ __forceinline__ void conv_layer(
    const _Float16* __restrict__ wfrag, const float* __restrict__ bias,
    const _Float16* src, void* dstv, int lane, int wave, int nwaves)
{
  constexpr int KHW    = KH * KW;
  constexpr int P      = Hd * Wd;
  constexpr int Ptiles = (P + 15) / 16;
  const int n      = lane & 15;
  const int hi     = lane >> 4;
  const int kbaseB = hi << 4;
  const int mrow   = hi << 3;

  for (int pt = wave; pt < Ptiles; pt += nwaves) {
    const int p      = pt * 16 + n;
    const bool pvalid = (p < P);
    const int pc     = pvalid ? p : (P - 1);     // clamp: lanes past P compute
    const int h      = pc / Wd;                  // garbage that is never stored
    const int w      = pc - h * Wd;

    v8f acc[MT];
#pragma unroll
    for (int mt = 0; mt < MT; ++mt)
#pragma unroll
      for (int r = 0; r < 8; ++r) {
        int co = mt * 16 + mrow + r;
        acc[mt][r] = (co < Cout) ? bias[co] : 0.0f;
      }

    const _Float16* wl0 = wfrag + (lane << 4);
    for (int kc = 0; kc < KCH; ++kc) {
      v16h bf;
#pragma unroll
      for (int e = 0; e < 16; ++e) {
        int kg = (kc << 5) + kbaseB + e;
        int ci = kg / KHW;
        int rr = kg - ci * KHW;
        int kh = rr / KW;
        int kw = rr - kh * KW;
        bf[e] = src[(ci * SH + h + kh) * SW + (w + kw)];
      }
#pragma unroll
      for (int mt = 0; mt < MT; ++mt) {
        v16h a = *(const v16h*)(wl0 + (((size_t)(mt * KCH + kc)) << 9));
        acc[mt] = __builtin_amdgcn_wmma_f32_16x16x32_f16(
            false, a, false, bf, (short)0, acc[mt], false, false);
      }
    }

#pragma unroll
    for (int mt = 0; mt < MT; ++mt)
#pragma unroll
      for (int r = 0; r < 8; ++r) {
        int co = mt * 16 + mrow + r;
        if (co < Cout && pvalid) {
          float v = acc[mt][r];
          if (RELU && v < 0.0f) v = 0.0f;
          if (OUTF32) ((float*)dstv)[co * P + p] = v;
          else ((_Float16*)dstv)[(co * (Hd + 2 * DHALO) + h + DHALO) *
                                 (Wd + 2 * DHALO) + (w + DHALO)] = (_Float16)v;
        }
      }
  }
}

// ---------------------------------------------------------------------------
// conv3 (64ch 24x11 -> 128ch 13x3, 12x9 kernel, K=6912): K-split across waves,
// all 8 co-tiles in registers, partials reduced with ds_add_f32 into a
// bias-preloaded f32 LDS buffer. Keeps all waves busy (only 3 spatial tiles).
// ---------------------------------------------------------------------------
__device__ __forceinline__ void conv3_ksplit(
    const _Float16* __restrict__ wfrag, const _Float16* src, float* dstf,
    int lane, int wave, int nwaves)
{
  constexpr int KCH = 216, MT = 8, P = 39, Wd = 3, SH = 24, SW = 11;
  constexpr int KHW = 108, KW = 9;
  const int n      = lane & 15;
  const int hi     = lane >> 4;
  const int kbaseB = hi << 4;
  const int mrow   = hi << 3;

  for (int pt = 0; pt < 3; ++pt) {
    const int p      = pt * 16 + n;
    const bool pvalid = (p < P);
    const int pc     = pvalid ? p : (P - 1);
    const int h      = pc / Wd;
    const int w      = pc - h * Wd;

    v8f acc[MT];
#pragma unroll
    for (int mt = 0; mt < MT; ++mt)
#pragma unroll
      for (int r = 0; r < 8; ++r) acc[mt][r] = 0.0f;

    const _Float16* wl0 = wfrag + (lane << 4);
    for (int kc = wave; kc < KCH; kc += nwaves) {
      v16h bf;
#pragma unroll
      for (int e = 0; e < 16; ++e) {
        int kg = (kc << 5) + kbaseB + e;
        int ci = kg / KHW;
        int rr = kg - ci * KHW;
        int kh = rr / KW;
        int kw = rr - kh * KW;
        bf[e] = src[(ci * SH + h + kh) * SW + (w + kw)];
      }
#pragma unroll
      for (int mt = 0; mt < MT; ++mt) {
        v16h a = *(const v16h*)(wl0 + (((size_t)(mt * KCH + kc)) << 9));
        acc[mt] = __builtin_amdgcn_wmma_f32_16x16x32_f16(
            false, a, false, bf, (short)0, acc[mt], false, false);
      }
    }

#pragma unroll
    for (int mt = 0; mt < MT; ++mt)
#pragma unroll
      for (int r = 0; r < 8; ++r) {
        int co = mt * 16 + mrow + r;
        if (pvalid) atomicAdd(&dstf[co * P + p], acc[mt][r]);
      }
  }
}

template<int C, int SH, int SW, int SHALO, int Hd, int Wd>
__device__ __forceinline__ void maxpool21(const _Float16* src, _Float16* dst,
                                          int tid, int nthreads) {
  constexpr int total = C * Hd * Wd;
  for (int i = tid; i < total; i += nthreads) {
    int w = i % Wd;
    int r = i / Wd;
    int h = r % Hd;
    int c = r / Hd;
    float a = (float)src[(c * SH + 2 * h + SHALO)     * SW + (w + SHALO)];
    float b = (float)src[(c * SH + 2 * h + 1 + SHALO) * SW + (w + SHALO)];
    dst[i] = (_Float16)((a > b) ? a : b);
  }
}

// ---------------------------------------------------------------------------
// Megakernel: one workgroup per window, 8 waves.
// LDS: regA | regB, each 32x106x17 f16 = 115,328 B (total 230,656 B).
// ---------------------------------------------------------------------------
__global__ __launch_bounds__(256, 1) void window_cnn(
    const float* __restrict__ x, const _Float16* __restrict__ wsw,
    const float* __restrict__ b1a, const float* __restrict__ b1b,
    const float* __restrict__ b1c, const float* __restrict__ b1d,
    const float* __restrict__ b2a, const float* __restrict__ b2b,
    const float* __restrict__ b3,  const float* __restrict__ bl,
    float* __restrict__ out)
{
  extern __shared__ char smem[];
  _Float16* regA = (_Float16*)smem;
  _Float16* regB = regA + HBUF;

  const int tid    = threadIdx.x;
  const int lane   = tid & 31;
  const int wave   = tid >> 5;
  const int nwaves = blockDim.x >> 5;
  const int bt     = blockIdx.x;
  const int b      = bt / NT;
  const int t      = bt - b * NT;

  // Zero both LDS buffers (halos must be 0; also legalizes clamped gathers).
  {
    uint4* z = (uint4*)smem;
    const int nz = (2 * HBUF * 2) / 16;          // 14,416 uint4
    for (int i = tid; i < nz; i += blockDim.x) z[i] = uint4{0, 0, 0, 0};
  }
  __syncthreads();

  // Window gather: x[b, t-7..t+7, :] -> regB interior of halo image [106][17]
  for (int i = tid; i < NF * NWIN; i += blockDim.x) {
    int f  = i / NWIN;
    int j  = i - f * NWIN;
    int tt = t - CTX + j;
    float v = (tt >= 0 && tt < NT) ? x[((size_t)(b * NT + tt)) * NF + f] : 0.0f;
    regB[(f + 1) * 17 + (j + 1)] = (_Float16)v;
  }
  __syncthreads();

  //            KCH  SH  SW  KH KW  Co   Hd  Wd  DH MT  relu f32
  conv_layer<    1, 106, 17, 3, 3, 32, 104, 15, 1, 2, true, false>(
      wsw + OFF1A, b1a, regB, regA, lane, wave, nwaves);
  __syncthreads();
  conv_layer<    9, 106, 17, 3, 3, 32, 104, 15, 1, 2, true, false>(
      wsw + OFF1B, b1b, regA, regB, lane, wave, nwaves);
  __syncthreads();
  conv_layer<    9, 106, 17, 3, 3, 32, 104, 15, 1, 2, true, false>(
      wsw + OFF1C, b1c, regB, regA, lane, wave, nwaves);
  __syncthreads();
  conv_layer<    9, 106, 17, 3, 3, 32, 104, 15, 1, 2, true, false>(
      wsw + OFF1D, b1d, regA, regB, lane, wave, nwaves);
  __syncthreads();
  maxpool21<32, 106, 17, 1, 52, 15>(regB, regA, tid, blockDim.x);  // 32x52x15
  __syncthreads();
  conv_layer<    9,  52, 15, 3, 3, 64,  50, 13, 0, 4, true, false>(
      wsw + OFF2A, b2a, regA, regB, lane, wave, nwaves);
  __syncthreads();
  conv_layer<   18,  50, 13, 3, 3, 64,  48, 11, 0, 4, true, false>(
      wsw + OFF2B, b2b, regB, regA, lane, wave, nwaves);
  __syncthreads();
  maxpool21<64, 48, 11, 0, 24, 11>(regA, regB, tid, blockDim.x);   // 64x24x11
  __syncthreads();

  // conv3: bias-init f32 accumulator in regA, K-split WMMA, relu+cvt to f16
  float*    c3acc = (float*)regA;
  _Float16* c3out = regA + C3OUT_OFF;            // byte ranges disjoint
  for (int i = tid; i < C3ACC_F32; i += blockDim.x) c3acc[i] = b3[i / 39];
  __syncthreads();
  conv3_ksplit(wsw + OFF3, regB, c3acc, lane, wave, nwaves);
  __syncthreads();
  for (int i = tid; i < C3ACC_F32; i += blockDim.x) {
    float v = c3acc[i];
    c3out[i] = (_Float16)((v > 0.0f) ? v : 0.0f);
  }
  __syncthreads();

  // 1x1 head (25ch), f32 out into regB: [25][39]
  conv_layer<    4,  13,  3, 1, 1, NOUT, 13, 3, 0, 2, false, true>(
      wsw + OFFL, bl, c3out, (void*)regB, lane, wave, nwaves);
  __syncthreads();

  // Final: mean over 13x3, /39
  if (tid < NOUT) {
    const float* o = (const float*)regB;
    float s = 0.0f;
#pragma unroll
    for (int p = 0; p < 39; ++p) s += o[tid * 39 + p];
    out[(size_t)bt * NOUT + tid] = s * (1.0f / 39.0f);
  }
}

// ---------------------------------------------------------------------------
extern "C" void kernel_launch(void* const* d_in, const int* in_sizes, int n_in,
                              void* d_out, int out_size, void* d_ws, size_t ws_size,
                              hipStream_t stream) {
  (void)in_sizes; (void)n_in; (void)out_size; (void)ws_size;

  const float* x   = (const float*)d_in[0];
  const float* w1a = (const float*)d_in[1];  const float* b1a = (const float*)d_in[2];
  const float* w1b = (const float*)d_in[3];  const float* b1b = (const float*)d_in[4];
  const float* w1c = (const float*)d_in[5];  const float* b1c = (const float*)d_in[6];
  const float* w1d = (const float*)d_in[7];  const float* b1d = (const float*)d_in[8];
  const float* w2a = (const float*)d_in[9];  const float* b2a = (const float*)d_in[10];
  const float* w2b = (const float*)d_in[11]; const float* b2b = (const float*)d_in[12];
  const float* w3  = (const float*)d_in[13]; const float* b3  = (const float*)d_in[14];
  const float* wl  = (const float*)d_in[15]; const float* bl  = (const float*)d_in[16];
  float* out = (float*)d_out;

  _Float16* ws16 = (_Float16*)d_ws;

  auto prep = [&](const float* w, unsigned off, int Cout, int Ktot,
                  int Mtiles, int Kchunks) {
    int total = Mtiles * Kchunks * 512;
    int grid  = (total + 255) / 256;
    prep_weights<<<grid, 256, 0, stream>>>(w, ws16 + off, Cout, Ktot,
                                           Mtiles, Kchunks);
  };
  prep(w1a, OFF1A, 32,    9, 2,   1);
  prep(w1b, OFF1B, 32,  288, 2,   9);
  prep(w1c, OFF1C, 32,  288, 2,   9);
  prep(w1d, OFF1D, 32,  288, 2,   9);
  prep(w2a, OFF2A, 64,  288, 4,   9);
  prep(w2b, OFF2B, 64,  576, 4,  18);
  prep(w3,  OFF3, 128, 6912, 8, 216);
  prep(wl,  OFFL, NOUT, 128, 2,   4);

  const size_t shmem = (size_t)2 * HBUF * sizeof(_Float16);  // 230,656 B
  window_cnn<<<NB * NT, 256, shmem, stream>>>(
      x, ws16, b1a, b1b, b1c, b1d, b2a, b2b, b3, bl, out);
}
